// ResidualVQVAE_33578054320521
// MI455X (gfx1250) — matrix operand order repaired and checked by reading, not compiled, not run on hardware
//
#include <hip/hip_runtime.h>
#include <hip/hip_bf16.h>

// ---------------------------------------------------------------------------
// Residual VQ-VAE forward for gfx1250 (MI455X). Memory-bound pipeline:
// NHWC layouts, fused elementwise stages, deterministic tree reductions,
// WMMA (f32<-f16 16x16x32) for the GEMM-shaped conv2 (K=256), TDM async
// weight staging into LDS (tensor_load_to_lds + s_wait_tensorcnt).
// ---------------------------------------------------------------------------

typedef __attribute__((ext_vector_type(16))) _Float16 v16h;
typedef __attribute__((ext_vector_type(8)))  _Float16 v8h;
typedef __attribute__((ext_vector_type(8)))  float    v8f;
typedef __attribute__((ext_vector_type(4)))  unsigned v4u;
typedef __attribute__((ext_vector_type(4)))  int      v4i;
typedef __attribute__((ext_vector_type(8)))  int      v8i_t;

#if __has_include(<hip/amd_detail/amd_gfx1250_TDM.h>)
#define TDM_6ARG 1   /* therock headers -> clang-23 6-arg builtin */
#else
#define TDM_6ARG 0   /* ROCm 7.2 clang-22 5-arg builtin */
#endif

#define BN_EPS 1e-5f

#define BATCH 64
#define H0 256
#define W0 256
#define H1 128
#define W1 128
#define H2 64
#define W2 64
#define NPIX1 (BATCH*H1*W1)   /* 1,048,576 : pixels of 16-ch tensors  */
#define NPIX2 (BATCH*H2*W2)   /*   262,144 : pixels of 4-ch tensors   */
#define NOUT  (BATCH*H0*W0)   /* 4,194,304 : output pixels            */

// ---------------------------------------------------------------------------
// TDM: async 1-D tile load (nelem f32, contiguous) global -> LDS offset.
// D# per CDNA5 ISA §8.3-8.6. Issue from one wave; EXEC is ignored by TDM so a
// wave-uniform branch is the correct gate. Completion via TENSORcnt.
// ---------------------------------------------------------------------------
__device__ inline void tdm_load_f32_1d(const float* src, unsigned lds_off,
                                       unsigned nelem)
{
  unsigned long long ga = (unsigned long long)(size_t)src;
  unsigned galo = __builtin_amdgcn_readfirstlane((unsigned)ga);
  unsigned gahi = __builtin_amdgcn_readfirstlane((unsigned)(ga >> 32));
  unsigned loff = __builtin_amdgcn_readfirstlane(lds_off);

  v4u g0;
  g0[0] = 1u;                                  // count=1, user mode
  g0[1] = loff;                                // lds_addr (bytes)
  g0[2] = galo;                                // global_addr[31:0]
  g0[3] = (gahi & 0x01ffffffu) | (2u << 30);   // global_addr[56:32] | type=2

  v8i_t g1;
  g1[0] = (2 << 16);                           // data_size=4B; wg_mask=0
  g1[1] = (int)(nelem << 16);                  // tensor_dim0[15:0] @ bits63:48
  g1[2] = (int)((nelem >> 16) | (1u << 16));   // tdim0 hi | tensor_dim1=1
  g1[3] = (int)(nelem << 16);                  // tile_dim0 @ bits127:112
  g1[4] = 1;                                   // tile_dim1=1, tile_dim2=0
  g1[5] = (int)nelem;                          // tensor_dim0_stride lo
  g1[6] = 0;                                   // stride hi | tdim1_stride lo
  g1[7] = 0;

  v4i g2; g2[0] = 1; g2[1] = 0; g2[2] = 0; g2[3] = 0;  // tensor_dim2=1
  v4i g3; g3[0] = 0; g3[1] = 0; g3[2] = 0; g3[3] = 0;

#if TDM_6ARG
  v8i_t gz; for (int i = 0; i < 8; ++i) gz[i] = 0;
  __builtin_amdgcn_tensor_load_to_lds(g0, g1, g2, g3, gz, 0);
#else
  __builtin_amdgcn_tensor_load_to_lds(g0, g1, g2, g3, 0);
#endif
}

// ---------------------------------------------------------------------------
// conv1: x (N,1,256,256) -> y (NPIX1 x 16) NHWC, k=4 s=2 p=1
// ---------------------------------------------------------------------------
__global__ __launch_bounds__(256) void conv1_kernel(
    const float* __restrict__ x, const float* __restrict__ w,
    const float* __restrict__ bias, float* __restrict__ y)
{
  __shared__ float wl[256];                     // wl[tap*16+co] = w[co*16+tap]
  int tid = threadIdx.x;
  wl[tid] = w[(tid & 15) * 16 + (tid >> 4)];
  __syncthreads();

  int p = blockIdx.x * 256 + tid;
  if (p >= NPIX1) return;
  int n = p >> 14, rem = p & 16383, oh = rem >> 7, ow = rem & 127;

  float acc[16];
#pragma unroll
  for (int co = 0; co < 16; ++co) acc[co] = bias[co];

  const float* xn = x + (size_t)n * (H0 * W0);
#pragma unroll
  for (int kh = 0; kh < 4; ++kh) {
    int ih = 2 * oh - 1 + kh;
    if (ih < 0 || ih >= H0) continue;
#pragma unroll
    for (int kw = 0; kw < 4; ++kw) {
      int iw = 2 * ow - 1 + kw;
      if (iw < 0 || iw >= W0) continue;
      float v = xn[ih * W0 + iw];
      const float* wt = &wl[(kh * 4 + kw) * 16];
#pragma unroll
      for (int co = 0; co < 16; ++co) acc[co] = fmaf(v, wt[co], acc[co]);
    }
  }
  float* yp = y + (size_t)p * 16;
#pragma unroll
  for (int co = 0; co < 16; ++co) yp[co] = acc[co];
}

// ---------------------------------------------------------------------------
// BatchNorm (training-mode batch stats): two-stage deterministic reduction.
// ---------------------------------------------------------------------------
template <int C>
__global__ __launch_bounds__(256) void bn_stats_kernel(
    const float* __restrict__ x, int npix, float* __restrict__ partials)
{
  float s[C], q[C];
#pragma unroll
  for (int c = 0; c < C; ++c) { s[c] = 0.f; q[c] = 0.f; }

  int stride = gridDim.x * blockDim.x;
  for (int p = blockIdx.x * blockDim.x + threadIdx.x; p < npix; p += stride) {
    const float* px = x + (size_t)p * C;
    __builtin_prefetch(px + (size_t)stride * C, 0, 1);   // global_prefetch_b8
#pragma unroll
    for (int c = 0; c < C; ++c) { float v = px[c]; s[c] += v; q[c] = fmaf(v, v, q[c]); }
  }

  __shared__ float red[256];
  int t = threadIdx.x;
#pragma unroll
  for (int c = 0; c < C; ++c) {
    red[t] = s[c]; __syncthreads();
    for (int off = 128; off > 0; off >>= 1) { if (t < off) red[t] += red[t + off]; __syncthreads(); }
    if (t == 0) partials[blockIdx.x * (2 * C) + c] = red[0];
    __syncthreads();
    red[t] = q[c]; __syncthreads();
    for (int off = 128; off > 0; off >>= 1) { if (t < off) red[t] += red[t + off]; __syncthreads(); }
    if (t == 0) partials[blockIdx.x * (2 * C) + C + c] = red[0];
    __syncthreads();
  }
}

template <int C>
__global__ __launch_bounds__(256) void bn_finalize_kernel(
    const float* __restrict__ partials, int nblk,
    const float* __restrict__ g, const float* __restrict__ beta,
    float count, float* __restrict__ ss)  // ss[0..C)=scale, ss[C..2C)=shift
{
  __shared__ float sums[2 * C];
  int t = threadIdx.x;
  if (t < 2 * C) {
    float s = 0.f;
    for (int b = 0; b < nblk; ++b) s += partials[b * (2 * C) + t];
    sums[t] = s;
  }
  __syncthreads();
  if (t < C) {
    float mean = sums[t] / count;
    float var  = sums[C + t] / count - mean * mean;
    float sc   = g[t] * rsqrtf(var + BN_EPS);
    ss[t]     = sc;
    ss[C + t] = fmaf(-mean, sc, beta[t]);
  }
}

template <int C>
__global__ __launch_bounds__(256) void bn_apply_relu_kernel(
    float* __restrict__ x, int n, const float* __restrict__ ss,
    float* __restrict__ copy)
{
  __shared__ float sl[2 * C];
  if (threadIdx.x < 2 * C) sl[threadIdx.x] = ss[threadIdx.x];
  __syncthreads();
  int stride = gridDim.x * blockDim.x;
  for (int i = blockIdx.x * blockDim.x + threadIdx.x; i < n; i += stride) {
    int c = i & (C - 1);
    float v = fmaxf(fmaf(x[i], sl[c], sl[C + c]), 0.f);
    x[i] = v;
    if (copy) copy[i] = v;
  }
}

// ---------------------------------------------------------------------------
// conv2 via WMMA: act1 (NPIX1 x 16, NHWC) -> y2 (NPIX2 x 4, NHWC)
// GEMM: M=16 (4 real out-ch, zero-padded), K=256 (16 taps x 16 ch), N=16 pos.
// Weights arrive by TDM async DMA into LDS (overlapped with im2col staging);
// A fragments prebuilt in WMMA register order (contiguous 32B/lane ->
// ds_load_b128), B rows padded to 528B stride (16B aligned, conflict-free).
// ---------------------------------------------------------------------------
__global__ __launch_bounds__(128) void conv2_wmma_kernel(
    const float* __restrict__ act1, const float* __restrict__ w,
    const float* __restrict__ bias, float* __restrict__ y2)
{
  __shared__ float wstage[1024];                        // TDM destination
  __shared__ __align__(32) _Float16 afrag[8][32][16];   // [kstep][lane][elem]
  __shared__ __align__(16) _Float16 plds[4][16][264];   // [wave][pos][k]

  const int tid  = threadIdx.x;
  const int wave = tid >> 5, lane = tid & 31;
  const int posBase = blockIdx.x * 64 + wave * 16;

  // 1) async DMA of the 4KB weight block into LDS (wave 0 only, wave-uniform)
  if (tid < 32) {
    tdm_load_f32_1d(w, (unsigned)(size_t)(void*)wstage, 1024u);
  }

  // 2) overlap: stage im2col patches (f32 -> f16) for this wave's 16 positions
  for (int idx = lane; idx < 16 * 256; idx += 32) {
    int j = idx >> 8, k = idx & 255;
    int ci = k & 15, tap = k >> 4, kh = tap >> 2, kw = tap & 3;
    int p = posBase + j;
    int n = p >> 12, rem = p & 4095, oh = rem >> 6, ow = rem & 63;
    int ih = 2 * oh - 1 + kh, iw = 2 * ow - 1 + kw;
    float v = 0.0f;
    if (ih >= 0 && ih < H1 && iw >= 0 && iw < W1)
      v = act1[(((size_t)(n * H1 + ih) * W1 + iw) << 4) + ci];
    plds[wave][j][k] = (_Float16)v;
  }

  // 3) wait for DMA, make wstage + plds visible block-wide
  if (tid < 32) __builtin_amdgcn_s_wait_tensorcnt(0);
  __syncthreads();

  // 4) build A fragments in register order: afrag[s][lane][e] = A[m][K(e)]
  //    lane<16: m=lane, K = 32s + {e<8 ? e : 16+e-8}; lane>=16: +8 on K base.
  for (int idx = tid; idx < 8 * 32 * 16; idx += 128) {
    int e = idx & 15, ln = (idx >> 4) & 31, s = idx >> 9;
    int m = ln & 15, hl2 = ln >> 4;
    int kk = s * 32 + hl2 * 8 + (e & 7) + ((e >> 3) << 4);
    int ci = kk & 15, tap = kk >> 4, kh = tap >> 2, kw = tap & 3;
    float v = (m < 4) ? wstage[((m * 16 + ci) * 4 + kh) * 4 + kw] : 0.0f;
    afrag[s][ln][e] = (_Float16)v;
  }
  __syncthreads();

  v8f acc = {};
  const int hl = lane >> 4;   // lane half selects K sub-ranges per ISA layout
  const int l  = lane & 15;
#pragma unroll
  for (int s = 0; s < 8; ++s) {
    const int kb = s * 32;
    v16h a = *(const v16h*)&afrag[s][lane][0];
    // B 32x16 f16: lanes 0-15 col=l K[kb+0..7,kb+16..23]; lanes 16-31 col=l K[kb+8..15,kb+24..31]
    v8h b0 = *(const v8h*)&plds[wave][l][kb + hl * 8];
    v8h b1 = *(const v8h*)&plds[wave][l][kb + 16 + hl * 8];
    v16h bm = __builtin_shufflevector(b0, b1, 0, 1, 2, 3, 4, 5, 6, 7,
                                      8, 9, 10, 11, 12, 13, 14, 15);
    acc = __builtin_amdgcn_wmma_f32_16x16x32_f16(
        /*neg_a=*/false, a, /*neg_b=*/false, bm,
        /*c_mod=*/(short)0, acc, /*reuse_a=*/false, /*reuse_b=*/false);
  }

  // D: VGPR r, lanes 0-15 hold M=r, N=lane. Need M=0..3 only.
  if (lane < 16) {
    int p = posBase + lane;
#pragma unroll
    for (int co = 0; co < 4; ++co)
      y2[(size_t)p * 4 + co] = acc[co] + bias[co];
  }
}

// ---------------------------------------------------------------------------
// Fused VQ block: z = Win*r + bin; nearest of 3 codebook vecs; loss partials;
// r -= Wout*qz + bout. All 1x1 convs on 4-channel NHWC pixels.
// ---------------------------------------------------------------------------
__global__ __launch_bounds__(256) void quant_kernel(
    float* __restrict__ r,
    const float* __restrict__ w_in, const float* __restrict__ b_in,
    const float* __restrict__ cb, const float* __restrict__ w_out,
    const float* __restrict__ b_out, float* __restrict__ partials, int npix)
{
  float wi[8], bi[2], c6[6], wo[8], bo[4];
#pragma unroll
  for (int i = 0; i < 8; ++i) wi[i] = w_in[i];
#pragma unroll
  for (int i = 0; i < 2; ++i) bi[i] = b_in[i];
#pragma unroll
  for (int i = 0; i < 6; ++i) c6[i] = cb[i];
#pragma unroll
  for (int i = 0; i < 8; ++i) wo[i] = w_out[i];
#pragma unroll
  for (int i = 0; i < 4; ++i) bo[i] = b_out[i];

  float lsum = 0.f;
  int stride = gridDim.x * blockDim.x;
  for (int p = blockIdx.x * blockDim.x + threadIdx.x; p < npix; p += stride) {
    float* rp = r + (size_t)p * 4;
    float r0 = rp[0], r1 = rp[1], r2 = rp[2], r3 = rp[3];
    float z0 = bi[0] + wi[0]*r0 + wi[1]*r1 + wi[2]*r2 + wi[3]*r3;
    float z1 = bi[1] + wi[4]*r0 + wi[5]*r1 + wi[6]*r2 + wi[7]*r3;

    float d0 = (z0-c6[0])*(z0-c6[0]) + (z1-c6[1])*(z1-c6[1]);
    float d1 = (z0-c6[2])*(z0-c6[2]) + (z1-c6[3])*(z1-c6[3]);
    float d2 = (z0-c6[4])*(z0-c6[4]) + (z1-c6[5])*(z1-c6[5]);
    float best = d0, q0 = c6[0], q1 = c6[1];          // argmin, first-occurrence
    if (d1 < best) { best = d1; q0 = c6[2]; q1 = c6[3]; }
    if (d2 < best) { best = d2; q0 = c6[4]; q1 = c6[5]; }

    float e0 = q0 - z0, e1 = q1 - z1;
    lsum += e0*e0 + e1*e1;

    rp[0] = r0 - (bo[0] + wo[0]*q0 + wo[1]*q1);
    rp[1] = r1 - (bo[1] + wo[2]*q0 + wo[3]*q1);
    rp[2] = r2 - (bo[2] + wo[4]*q0 + wo[5]*q1);
    rp[3] = r3 - (bo[3] + wo[6]*q0 + wo[7]*q1);
  }

  __shared__ float red[256];
  int t = threadIdx.x;
  red[t] = lsum; __syncthreads();
  for (int off = 128; off > 0; off >>= 1) { if (t < off) red[t] += red[t + off]; __syncthreads(); }
  if (t == 0) partials[blockIdx.x] = red[0];
}

// total_loss = 1.25 * (S0 + S1) / (NPIX2*2)   (codebook + COMMIT*commit, fwd-equal)
__global__ __launch_bounds__(256) void loss_finalize_kernel(
    const float* __restrict__ qp, float* __restrict__ out_loss)
{
  __shared__ float red[256];
  int t = threadIdx.x;
  red[t] = qp[t] + qp[t + 256] + qp[t + 512] + qp[t + 768];
  __syncthreads();
  for (int off = 128; off > 0; off >>= 1) { if (t < off) red[t] += red[t + off]; __syncthreads(); }
  if (t == 0) *out_loss = red[0] * (1.25f / (float)(NPIX2 * 2));
}

// ---------------------------------------------------------------------------
// convT1: (e - r) (NPIX2 x 4) -> y (NPIX1 x 16) NHWC; k=4 s=2 p=1 transpose conv
// w layout (ci=4, co=16, kh, kw). oh = 2*ih - 1 + kh.
// ---------------------------------------------------------------------------
__global__ __launch_bounds__(256) void convt1_kernel(
    const float* __restrict__ e, const float* __restrict__ r,
    const float* __restrict__ w, const float* __restrict__ bias,
    float* __restrict__ y)
{
  __shared__ float wl[1024];
  for (int i = threadIdx.x; i < 1024; i += 256) wl[i] = w[i];
  __syncthreads();

  int p = blockIdx.x * 256 + threadIdx.x;
  if (p >= NPIX1) return;
  int n = p >> 14, rem = p & 16383, oh = rem >> 7, ow = rem & 127;

  float acc[16];
#pragma unroll
  for (int co = 0; co < 16; ++co) acc[co] = bias[co];

  int kh0 = (oh + 1) & 1, kw0 = (ow + 1) & 1;
#pragma unroll
  for (int dkh = 0; dkh < 2; ++dkh) {
    int kh = kh0 + 2 * dkh;
    int ih = (oh + 1 - kh) >> 1;
    if (ih < 0 || ih >= H2) continue;
#pragma unroll
    for (int dkw = 0; dkw < 2; ++dkw) {
      int kw = kw0 + 2 * dkw;
      int iw = (ow + 1 - kw) >> 1;
      if (iw < 0 || iw >= W2) continue;
      int base = ((n * H2 + ih) * W2 + iw) * 4;
#pragma unroll
      for (int ci = 0; ci < 4; ++ci) {
        float v = e[base + ci] - r[base + ci];
#pragma unroll
        for (int co = 0; co < 16; ++co)
          acc[co] = fmaf(v, wl[ci * 256 + co * 16 + kh * 4 + kw], acc[co]);
      }
    }
  }
  float* yp = y + (size_t)p * 16;
#pragma unroll
  for (int co = 0; co < 16; ++co) yp[co] = acc[co];
}

// ---------------------------------------------------------------------------
// convT2 + sigmoid: d1 (NPIX1 x 16) -> out (N,1,256,256). w (ci=16,1,4,4).
// ---------------------------------------------------------------------------
__global__ __launch_bounds__(256) void convt2_sigmoid_kernel(
    const float* __restrict__ din, const float* __restrict__ w,
    const float* __restrict__ bias, float* __restrict__ out)
{
  __shared__ float wl[256];
  wl[threadIdx.x] = w[threadIdx.x];
  __syncthreads();

  int p = blockIdx.x * 256 + threadIdx.x;
  if (p >= NOUT) return;
  int n = p >> 16, rem = p & 65535, oh = rem >> 8, ow = rem & 255;

  float acc = bias[0];
  int kh0 = (oh + 1) & 1, kw0 = (ow + 1) & 1;
#pragma unroll
  for (int dkh = 0; dkh < 2; ++dkh) {
    int kh = kh0 + 2 * dkh;
    int ih = (oh + 1 - kh) >> 1;
    if (ih < 0 || ih >= H1) continue;
#pragma unroll
    for (int dkw = 0; dkw < 2; ++dkw) {
      int kw = kw0 + 2 * dkw;
      int iw = (ow + 1 - kw) >> 1;
      if (iw < 0 || iw >= W1) continue;
      const float* dp = din + ((size_t)(n * H1 + ih) * W1 + iw) * 16;
#pragma unroll
      for (int ci = 0; ci < 16; ++ci)
        acc = fmaf(dp[ci], wl[ci * 16 + kh * 4 + kw], acc);
    }
  }
  out[p] = 1.0f / (1.0f + __expf(-acc));
}

// ---------------------------------------------------------------------------
// Launch. d_in order = setup_inputs() insertion order:
//  0:x 1:e1w 2:e1b 3:bn1g 4:bn1b 5:e2w 6:e2b 7:bn2g 8:bn2b 9:d1w 10:d1b
//  11:bn3g 12:bn3b 13:d2w 14:d2b 15..19:q0{w_in,b_in,codebook,w_out,b_out}
//  20..24:q1{...}. Output: [d (4,194,304 f32), total_loss (1 f32)].
// ---------------------------------------------------------------------------
extern "C" void kernel_launch(void* const* d_in, const int* in_sizes, int n_in,
                              void* d_out, int out_size, void* d_ws, size_t ws_size,
                              hipStream_t stream)
{
  (void)in_sizes; (void)n_in; (void)out_size; (void)ws_size;
  const float* x    = (const float*)d_in[0];
  const float* e1w  = (const float*)d_in[1];
  const float* e1b  = (const float*)d_in[2];
  const float* bn1g = (const float*)d_in[3];
  const float* bn1b = (const float*)d_in[4];
  const float* e2w  = (const float*)d_in[5];
  const float* e2b  = (const float*)d_in[6];
  const float* bn2g = (const float*)d_in[7];
  const float* bn2b = (const float*)d_in[8];
  const float* d1w  = (const float*)d_in[9];
  const float* d1b  = (const float*)d_in[10];
  const float* bn3g = (const float*)d_in[11];
  const float* bn3b = (const float*)d_in[12];
  const float* d2w  = (const float*)d_in[13];
  const float* d2b  = (const float*)d_in[14];

  float* ws    = (float*)d_ws;
  float* act1  = ws;                               // NPIX1*16
  float* ebuf  = act1 + (size_t)NPIX1 * 16;        // NPIX2*4
  float* rbuf  = ebuf + (size_t)NPIX2 * 4;         // NPIX2*4
  float* d1buf = rbuf + (size_t)NPIX2 * 4;         // NPIX1*16
  float* part  = d1buf + (size_t)NPIX1 * 16;       // 256*32
  float* ss1   = part + 256 * 32;                  // 32
  float* ss2   = ss1 + 32;                         // 8
  float* ss3   = ss2 + 8;                          // 32
  float* qpart = ss3 + 32;                         // 1024 (2 x 512)

  float* dout = (float*)d_out;
  float* loss = dout + (size_t)NOUT;

  // encoder
  conv1_kernel<<<NPIX1 / 256, 256, 0, stream>>>(x, e1w, e1b, act1);
  bn_stats_kernel<16><<<256, 256, 0, stream>>>(act1, NPIX1, part);
  bn_finalize_kernel<16><<<1, 256, 0, stream>>>(part, 256, bn1g, bn1b, (float)NPIX1, ss1);
  bn_apply_relu_kernel<16><<<4096, 256, 0, stream>>>(act1, NPIX1 * 16, ss1, nullptr);

  conv2_wmma_kernel<<<NPIX2 / 64, 128, 0, stream>>>(act1, e2w, e2b, ebuf);
  bn_stats_kernel<4><<<256, 256, 0, stream>>>(ebuf, NPIX2, part);
  bn_finalize_kernel<4><<<1, 256, 0, stream>>>(part, 256, bn2g, bn2b, (float)NPIX2, ss2);
  bn_apply_relu_kernel<4><<<2048, 256, 0, stream>>>(ebuf, NPIX2 * 4, ss2, rbuf);

  // residual quantizer blocks (in-place residual update, loss partials)
  for (int q = 0; q < 2; ++q) {
    const float* wi = (const float*)d_in[15 + q * 5 + 0];
    const float* bi = (const float*)d_in[15 + q * 5 + 1];
    const float* cb = (const float*)d_in[15 + q * 5 + 2];
    const float* wo = (const float*)d_in[15 + q * 5 + 3];
    const float* bo = (const float*)d_in[15 + q * 5 + 4];
    quant_kernel<<<512, 256, 0, stream>>>(rbuf, wi, bi, cb, wo, bo, qpart + q * 512, NPIX2);
  }
  loss_finalize_kernel<<<1, 256, 0, stream>>>(qpart, loss);

  // decoder
  convt1_kernel<<<NPIX1 / 256, 256, 0, stream>>>(ebuf, rbuf, d1w, d1b, d1buf);
  bn_stats_kernel<16><<<256, 256, 0, stream>>>(d1buf, NPIX1, part);
  bn_finalize_kernel<16><<<1, 256, 0, stream>>>(part, 256, bn3g, bn3b, (float)NPIX1, ss3);
  bn_apply_relu_kernel<16><<<4096, 256, 0, stream>>>(d1buf, NPIX1 * 16, ss3, nullptr);

  convt2_sigmoid_kernel<<<NOUT / 256, 256, 0, stream>>>(d1buf, d2w, d2b, dout);
}
